// Mamba2D_MIL_69097433858257
// MI455X (gfx1250) — matrix operand order repaired and checked by reading, not compiled, not run on hardware
//
#include <hip/hip_runtime.h>
#include <hip/hip_bf16.h>

// ---------------- types ----------------
typedef __bf16 bf16;
typedef __attribute__((ext_vector_type(16))) __bf16 v16bf;
typedef __attribute__((ext_vector_type(8)))  __bf16 v8bf;
typedef __attribute__((ext_vector_type(8)))  float  v8f;
typedef __attribute__((ext_vector_type(4)))  int    v4i;

#if __has_builtin(__builtin_amdgcn_global_load_async_to_lds_b128)
typedef __attribute__((address_space(1))) v4i* glob_b128_t;
typedef __attribute__((address_space(3))) v4i* lds_b128_t;
#define HAVE_ASYNC_LDS 1
#else
#define HAVE_ASYNC_LDS 0
#endif

// ---------------- model dims ----------------
constexpr int kNP   = 4096;   // tokens
constexpr int kIN   = 1024;   // input dim
constexpr int kDM   = 512;    // d_model
constexpr int kDI   = 1024;   // d_inner
constexpr int kDTR  = 32;     // dt_rank
constexpr int kNS   = 16;     // d_state
constexpr int kCONV = 4;      // conv width
constexpr int kHW   = 64;     // grid side (64x64 tokens)

constexpr int ACT_NONE = 0, ACT_RELU = 1, ACT_TANH = 2;

// ======================================================================
// f32 -> bf16 conversion
// ======================================================================
__global__ __launch_bounds__(256) void k_cvt_bf16(const float* __restrict__ s,
                                                  bf16* __restrict__ d, int n) {
  int i = blockIdx.x * 256 + threadIdx.x;
  if (i < n) d[i] = (bf16)s[i];
}

// ======================================================================
// WMMA GEMM: C[M x N] = act( A[M x K] * W[N x K]^T + bias[N] )
// A bf16 row-major (lda), W bf16 row-major (ldw >= K).
// Block = 256 threads = 8 waves; wave w owns a 32x64 output tile
// (rows blk.x*256 + w*32 .. +32, cols blk.y*64 .. +64) as 2x4 16x16 WMMA
// accumulators; each LDS B-fragment is reused for both A fragments.
// B tile (64 x 32 K) is staged in LDS with async loads.
// Requires: M%256==0, N%64==0, K%32==0 (true for every GEMM here).
// ======================================================================
__global__ __launch_bounds__(256) void k_gemm_bf16_wmma(
    const bf16* __restrict__ A, int lda,
    const bf16* __restrict__ W, int ldw,
    const float* __restrict__ bias,
    float* __restrict__ C, int ldc,
    bf16* __restrict__ Cb, int ldcb,
    int M, int N, int K, int act) {
  __shared__ __align__(64) bf16 lbs[64 * 32];  // 64 N-rows x 32 K, 4 KB

  const int wave  = threadIdx.x >> 5;
  const int lane  = threadIdx.x & 31;
  const int lhalf = (lane >> 4) & 1;  // 0: lanes 0-15, 1: lanes 16-31
  const int lmod  = lane & 15;
  const int m0    = blockIdx.x * 256 + wave * 32;
  const int n0    = blockIdx.y * 64;

  v8f acc[2][4] = {};

  // A fragment bases: lane row = m0(+16)+lmod; lhalf selects K sub-phase
  const bf16* aptr0 = A + (size_t)(m0 + lmod) * lda + lhalf * 16;
  const bf16* aptr1 = aptr0 + (size_t)16 * lda;

  for (int k0 = 0; k0 < K; k0 += 32) {
    // ---- stage B tile: rows n0..n0+63, cols k0..k0+31 (16B per thread) ----
    {
      const int r = threadIdx.x >> 2;        // 0..63
      const int c = (threadIdx.x & 3) * 8;   // 0,8,16,24
      const bf16* gsrc = W + (size_t)(n0 + r) * ldw + k0 + c;
      bf16* ldst = &lbs[r * 32 + c];
#if HAVE_ASYNC_LDS
      __builtin_amdgcn_global_load_async_to_lds_b128(
          (glob_b128_t)const_cast<bf16*>(gsrc), (lds_b128_t)ldst, 0, 0);
#else
      *(v8bf*)ldst = *(const v8bf*)gsrc;
#endif
    }
#if HAVE_ASYNC_LDS && __has_builtin(__builtin_amdgcn_s_wait_asynccnt)
    __builtin_amdgcn_s_wait_asynccnt(0);
#endif
    __syncthreads();

    // prefetch A two K-steps ahead (global_prefetch_b8, keep-near hint)
    if (k0 + 64 < K) {
      __builtin_prefetch(aptr0 + k0 + 64, 0, 3);
      __builtin_prefetch(aptr1 + k0 + 64, 0, 3);
    }

    // ---- A fragments (16x32 bf16 each):
    //      lane<16 -> K {0..15}, lane>=16 -> K {16..31}
    v8bf a0lo = *(const v8bf*)(aptr0 + k0);
    v8bf a0hi = *(const v8bf*)(aptr0 + k0 + 8);
    v8bf a1lo = *(const v8bf*)(aptr1 + k0);
    v8bf a1hi = *(const v8bf*)(aptr1 + k0 + 8);
    v16bf af0, af1;
#pragma unroll
    for (int i = 0; i < 8; ++i) {
      af0[i] = a0lo[i]; af0[8 + i] = a0hi[i];
      af1[i] = a1lo[i]; af1[8 + i] = a1hi[i];
    }

    // ---- 4 B fragments from LDS, each feeding 2 WMMAs ----
#pragma unroll
    for (int nt = 0; nt < 4; ++nt) {
      // B 32x16: lane<16 holds (N=lane, K 0..15); lane>=16 holds (N=lane-16, K 16..31)
      const bf16* bl = &lbs[(nt * 16 + lmod) * 32 + lhalf * 16];
      v16bf bfrag = *(const v16bf*)bl;
      acc[0][nt] = __builtin_amdgcn_wmma_f32_16x16x32_bf16(
          false, af0, false, bfrag, (short)0, acc[0][nt], false, false);
      acc[1][nt] = __builtin_amdgcn_wmma_f32_16x16x32_bf16(
          false, af1, false, bfrag, (short)0, acc[1][nt], false, false);
    }
    __syncthreads();  // lbs reused next K-step
  }

  // ---- epilogue: VGPR r of lane l holds D[M = r + 8*lhalf, N = lmod] ----
#pragma unroll
  for (int half = 0; half < 2; ++half) {
    const int row = m0 + half * 16 + lhalf * 8;
#pragma unroll
    for (int nt = 0; nt < 4; ++nt) {
      const int col = n0 + nt * 16 + lmod;
      const float bv = bias ? bias[col] : 0.0f;
#pragma unroll
      for (int r = 0; r < 8; ++r) {
        float v = acc[half][nt][r] + bv;
        if (act == ACT_RELU) v = v > 0.0f ? v : 0.0f;
        else if (act == ACT_TANH) v = tanhf(v);
        C[(size_t)(row + r) * ldc + col] = v;
        if (Cb) Cb[(size_t)(row + r) * ldcb + col] = (bf16)v;
      }
    }
  }
}

// ======================================================================
// causal depthwise conv (width 4) + SiLU. col_major selects token stride.
// ======================================================================
__global__ __launch_bounds__(256) void k_dwconv_silu(
    const float* __restrict__ xin, const float* __restrict__ w,
    const float* __restrict__ b, float* __restrict__ xc,
    bf16* __restrict__ xcb, int col_major) {
  int idx = blockIdx.x * 256 + threadIdx.x;
  if (idx >= kNP * kDI) return;
  const int d  = idx % kDI;
  const int mt = idx / kDI;           // memory token (h*64+w)
  const int a_ = mt / kHW, b_ = mt % kHW;
  const int s = col_major ? b_ : a_;  // sequence id
  const int t = col_major ? a_ : b_;  // time step
  float acc = b[d];
#pragma unroll
  for (int j = 0; j < kCONV; ++j) {
    const int tt = t - (kCONV - 1) + j;
    if (tt >= 0) {
      const int nmt = col_major ? tt * kHW + s : s * kHW + tt;
      acc += w[d * kCONV + j] * xin[(size_t)nmt * kDI + d];
    }
  }
  const float sv = acc / (1.0f + __expf(-acc));
  xc[(size_t)mt * kDI + d]  = sv;
  xcb[(size_t)mt * kDI + d] = (bf16)sv;
}

// ======================================================================
// selective scan, fused softplus(dt) and SiLU(z) gating.
// one thread per (sequence, channel); 64 serial steps x 16 states.
// ======================================================================
__global__ __launch_bounds__(256) void k_scan(
    const float* __restrict__ u, const float* __restrict__ dtr,
    const float* __restrict__ xdbl,  // NP x 64; B at +kDTR, C at +kDTR+kNS
    const float* __restrict__ A_log, const float* __restrict__ Dp,
    const float* __restrict__ z, bf16* __restrict__ yg, int col_major) {
  int idx = blockIdx.x * 256 + threadIdx.x;
  if (idx >= kHW * kDI) return;
  const int d = idx % kDI;
  const int s = idx / kDI;
  float A[kNS];
#pragma unroll
  for (int n = 0; n < kNS; ++n) A[n] = -__expf(A_log[d * kNS + n]);
  float h[kNS] = {};
  const float Dv = Dp[d];
  for (int t = 0; t < kHW; ++t) {
    const int mt = col_major ? t * kHW + s : s * kHW + t;
    const float ut  = u[(size_t)mt * kDI + d];
    float dtv = dtr[(size_t)mt * kDI + d];
    dtv = (dtv > 20.0f) ? dtv : __logf(1.0f + __expf(dtv));  // softplus
    const float* Bv = xdbl + (size_t)mt * 64 + kDTR;
    const float* Cv = Bv + kNS;
    const float du = dtv * ut;
    float y = 0.0f;
#pragma unroll
    for (int n = 0; n < kNS; ++n) {
      h[n] = __expf(dtv * A[n]) * h[n] + du * Bv[n];
      y += h[n] * Cv[n];
    }
    y += ut * Dv;
    const float zv = z[(size_t)mt * kDI + d];
    const float g  = zv / (1.0f + __expf(-zv));
    yg[(size_t)mt * kDI + d] = (bf16)(y * g);
  }
}

// ======================================================================
// residual add + LayerNorm(512). One wave per token.
// ======================================================================
__global__ __launch_bounds__(256) void k_add_ln(
    const float* __restrict__ xc, const float* __restrict__ xres,
    const float* __restrict__ g, const float* __restrict__ b,
    float* __restrict__ out, bf16* __restrict__ outb) {
  const int wv = (blockIdx.x * 256 + threadIdx.x) >> 5;
  const int lane = threadIdx.x & 31;
  if (wv >= kNP) return;
  const float* p = xc + (size_t)wv * kDM;
  const float* q = xres + (size_t)wv * kDM;
  float s = 0.0f, s2 = 0.0f;
  for (int i = lane; i < kDM; i += 32) {
    const float v = p[i] + q[i];
    s += v; s2 += v * v;
  }
#pragma unroll
  for (int off = 16; off; off >>= 1) { s += __shfl_down(s, off); s2 += __shfl_down(s2, off); }
  s = __shfl(s, 0); s2 = __shfl(s2, 0);
  const float mean = s / kDM;
  const float inv  = rsqrtf(s2 / kDM - mean * mean + 1e-5f);
  for (int i = lane; i < kDM; i += 32) {
    const float v = (p[i] + q[i] - mean) * inv * g[i] + b[i];
    out[(size_t)wv * kDM + i]  = v;
    outb[(size_t)wv * kDM + i] = (bf16)v;
  }
}

// ======================================================================
// attention: score per token = dot(tanh-projected 128, w2) + b2 (wave/token)
// ======================================================================
__global__ __launch_bounds__(256) void k_att_score(
    const float* __restrict__ t1, const float* __restrict__ w2,
    const float* __restrict__ b2, float* __restrict__ out) {
  const int wv = (blockIdx.x * 256 + threadIdx.x) >> 5;
  const int lane = threadIdx.x & 31;
  if (wv >= kNP) return;
  float s = 0.0f;
  for (int i = lane; i < 128; i += 32) s += t1[(size_t)wv * 128 + i] * w2[i];
#pragma unroll
  for (int off = 16; off; off >>= 1) s += __shfl_down(s, off);
  if (lane == 0) out[wv] = s + b2[0];
}

__global__ __launch_bounds__(1024) void k_softmax_np(float* __restrict__ a) {
  __shared__ float red[32];
  const int tid = threadIdx.x, lane = tid & 31, wv = tid >> 5;
  float m = -1e30f;
  for (int i = tid; i < kNP; i += 1024) m = fmaxf(m, a[i]);
#pragma unroll
  for (int off = 16; off; off >>= 1) m = fmaxf(m, __shfl_down(m, off));
  if (lane == 0) red[wv] = m;
  __syncthreads();
  if (wv == 0) {
    float m2 = red[lane];
#pragma unroll
    for (int off = 16; off; off >>= 1) m2 = fmaxf(m2, __shfl_down(m2, off));
    if (lane == 0) red[0] = m2;
  }
  __syncthreads();
  const float mx = red[0];
  __syncthreads();
  float s = 0.0f;
  for (int i = tid; i < kNP; i += 1024) { const float e = __expf(a[i] - mx); a[i] = e; s += e; }
#pragma unroll
  for (int off = 16; off; off >>= 1) s += __shfl_down(s, off);
  if (lane == 0) red[wv] = s;
  __syncthreads();
  if (wv == 0) {
    float s2 = red[lane];
#pragma unroll
    for (int off = 16; off; off >>= 1) s2 += __shfl_down(s2, off);
    if (lane == 0) red[0] = s2;
  }
  __syncthreads();
  const float inv = 1.0f / red[0];
  for (int i = tid; i < kNP; i += 1024) a[i] *= inv;
}

__global__ __launch_bounds__(512) void k_bag(
    const float* __restrict__ aw, const float* __restrict__ hf,
    float* __restrict__ bag) {
  const int d = threadIdx.x;
  float s = 0.0f;
  for (int t = 0; t < kNP; ++t) s += aw[t] * hf[(size_t)t * kDM + d];
  bag[d] = s;
}

__global__ __launch_bounds__(64) void k_logits(
    const float* __restrict__ bag, const float* __restrict__ Wc,
    const float* __restrict__ bc, float* __restrict__ out) {
  const int c = threadIdx.x >> 5, lane = threadIdx.x & 31;
  float s = 0.0f;
  for (int i = lane; i < kDM; i += 32) s += bag[i] * Wc[c * kDM + i];
#pragma unroll
  for (int off = 16; off; off >>= 1) s += __shfl_down(s, off);
  if (lane == 0) out[c] = s + bc[c];
}

// ======================================================================
// host orchestration
// ======================================================================
extern "C" void kernel_launch(void* const* d_in, const int* in_sizes, int n_in,
                              void* d_out, int out_size, void* d_ws, size_t ws_size,
                              hipStream_t stream) {
  (void)in_sizes; (void)n_in; (void)out_size; (void)ws_size;

  // ---- inputs (setup_inputs dict order, params flattened recursively) ----
  const float* x      = (const float*)d_in[0];
  const float* proj_W = (const float*)d_in[1];
  const float* proj_b = (const float*)d_in[2];
  // per-mamba params: 0 W_in, 1 conv_w, 2 conv_b, 3 W_x, 4 W_dt, 5 b_dt,
  //                   6 A_log, 7 D, 8 W_out ; layer block = row(9)+col(9)+ln_g+ln_b
  auto MP = [&](int l, int o, int k) -> const float* {
    return (const float*)d_in[3 + l * 20 + o * 9 + k];
  };
  auto LNG = [&](int l) -> const float* { return (const float*)d_in[3 + l * 20 + 18]; };
  auto LNB = [&](int l) -> const float* { return (const float*)d_in[3 + l * 20 + 19]; };
  const float* att_W1 = (const float*)d_in[43];
  const float* att_b1 = (const float*)d_in[44];
  const float* att_W2 = (const float*)d_in[45];
  const float* att_b2 = (const float*)d_in[46];
  const float* cls_W  = (const float*)d_in[47];
  const float* cls_b  = (const float*)d_in[48];
  float* out = (float*)d_out;

  // ---- workspace arena ----
  size_t off = 0;
  auto alloc = [&](size_t bytes) -> char* {
    char* p = (char*)d_ws + off;
    off = (off + bytes + 255) & ~(size_t)255;
    return p;
  };
  // bf16 buffers
  bf16* bX     = (bf16*)alloc((size_t)kNP * kIN * 2);
  bf16* bWproj = (bf16*)alloc((size_t)kDM * kIN * 2);
  bf16* bW1    = (bf16*)alloc((size_t)128 * kDM * 2);
  bf16* bWin[2][2]; bf16* bWx[2][2]; bf16* bWdt[2][2]; bf16* bWout[2][2];
  for (int l = 0; l < 2; ++l)
    for (int o = 0; o < 2; ++o) {
      bWin[l][o]  = (bf16*)alloc((size_t)2 * kDI * kDM * 2);
      bWx[l][o]   = (bf16*)alloc((size_t)64 * kDI * 2);
      bWdt[l][o]  = (bf16*)alloc((size_t)kDI * kDTR * 2);
      bWout[l][o] = (bf16*)alloc((size_t)kDM * kDI * 2);
    }
  bf16*  Hb    = (bf16*)alloc((size_t)kNP * kDM * 2);
  bf16*  XCB   = (bf16*)alloc((size_t)kNP * kDI * 2);
  bf16*  XDBLB = (bf16*)alloc((size_t)kNP * 64 * 2);
  bf16*  YG    = (bf16*)alloc((size_t)kNP * kDI * 2);
  bf16*  XRb   = (bf16*)alloc((size_t)kNP * kDM * 2);
  // f32 buffers
  float* Hf   = (float*)alloc((size_t)kNP * kDM * 4);
  float* XIN  = (float*)alloc((size_t)kNP * kDI * 4);  // reused as DTR buf & T1
  float* Z    = (float*)alloc((size_t)kNP * kDI * 4);  // reused as scores/bag
  float* XC   = (float*)alloc((size_t)kNP * kDI * 4);
  float* XDBL = (float*)alloc((size_t)kNP * 64 * 4);
  float* XRf  = (float*)alloc((size_t)kNP * kDM * 4);
  float* MOf  = (float*)alloc((size_t)kNP * kDM * 4);
  float* DTRb = XIN;  // xin dead after conv; same size

  auto cvt = [&](const float* s, bf16* d, int n) {
    k_cvt_bf16<<<(n + 255) / 256, 256, 0, stream>>>(s, d, n);
  };
  auto gemm = [&](const bf16* A, int lda, const bf16* W, int ldw,
                  const float* bias, float* C, int ldc, bf16* Cb, int ldcb,
                  int M, int N, int K, int act) {
    dim3 g(M / 256, N / 64);
    k_gemm_bf16_wmma<<<g, 256, 0, stream>>>(A, lda, W, ldw, bias, C, ldc,
                                            Cb, ldcb, M, N, K, act);
  };

  // ---- weight conversions (deterministic each call) ----
  cvt(x, bX, kNP * kIN);
  cvt(proj_W, bWproj, kDM * kIN);
  cvt(att_W1, bW1, 128 * kDM);
  for (int l = 0; l < 2; ++l)
    for (int o = 0; o < 2; ++o) {
      cvt(MP(l, o, 0), bWin[l][o], 2 * kDI * kDM);
      cvt(MP(l, o, 3), bWx[l][o], 64 * kDI);
      cvt(MP(l, o, 4), bWdt[l][o], kDI * kDTR);
      cvt(MP(l, o, 8), bWout[l][o], kDM * kDI);
    }

  // ---- proj + relu ----
  gemm(bX, kIN, bWproj, kIN, proj_b, Hf, kDM, Hb, kDM, kNP, kDM, kIN, ACT_RELU);

  // ---- mamba blocks ----
  auto mamba = [&](int l, int o, const bf16* inB, float* outF, bf16* outB) {
    const int cm = o;  // 0=row (stride 1 token), 1=col (stride 64 tokens)
    // xz = x @ W_in^T, split halves into XIN / Z
    gemm(inB, kDM, bWin[l][o], kDM, nullptr, XIN, kDI, nullptr, 0, kNP, kDI, kDM, ACT_NONE);
    gemm(inB, kDM, bWin[l][o] + (size_t)kDI * kDM, kDM, nullptr, Z, kDI, nullptr, 0,
         kNP, kDI, kDM, ACT_NONE);
    // causal depthwise conv + silu
    k_dwconv_silu<<<(kNP * kDI) / 256, 256, 0, stream>>>(
        XIN, MP(l, o, 1), MP(l, o, 2), XC, XCB, cm);
    // xdbl = xc @ W_x^T (N=64: [dt_rank | B | C])
    gemm(XCB, kDI, bWx[l][o], kDI, nullptr, XDBL, 64, XDBLB, 64, kNP, 64, kDI, ACT_NONE);
    // dt_raw = xdbl[:, :32] @ W_dt^T + b_dt   (softplus fused in scan)
    gemm(XDBLB, 64, bWdt[l][o], kDTR, MP(l, o, 5), DTRb, kDI, nullptr, 0,
         kNP, kDI, kDTR, ACT_NONE);
    // selective scan + SiLU(z) gating -> bf16
    k_scan<<<(kHW * kDI) / 256, 256, 0, stream>>>(
        XC, DTRb, XDBL, MP(l, o, 6), MP(l, o, 7), Z, YG, cm);
    // out = y_gated @ W_out^T
    gemm(YG, kDI, bWout[l][o], kDI, nullptr, outF, kDM, outB, kDM, kNP, kDM, kDI, ACT_NONE);
  };

  for (int l = 0; l < 2; ++l) {
    mamba(l, 0, Hb, XRf, XRb);            // row mamba
    mamba(l, 1, XRb, MOf, nullptr);       // col mamba
    k_add_ln<<<(kNP * 32) / 256, 256, 0, stream>>>(MOf, Hf, LNG(l), LNB(l), Hf, Hb);
  }

  // ---- attention pooling + classifier ----
  float* T1     = XIN;          // 4096 x 128
  float* SCORES = Z;            // 4096
  float* BAG    = Z + 8192;     // 512
  gemm(Hb, kDM, bW1, kDM, att_b1, T1, 128, nullptr, 0, kNP, 128, kDM, ACT_TANH);
  k_att_score<<<(kNP * 32) / 256, 256, 0, stream>>>(T1, att_W2, att_b2, SCORES);
  k_softmax_np<<<1, 1024, 0, stream>>>(SCORES);
  k_bag<<<1, 512, 0, stream>>>(SCORES, Hf, BAG);
  k_logits<<<1, 64, 0, stream>>>(BAG, cls_W, cls_b, out);
}
